// LocalWindowAttention_76132590289169
// MI455X (gfx1250) — matrix operand order repaired and checked by reading, not compiled, not run on hardware
//
#include <hip/hip_runtime.h>

typedef __attribute__((ext_vector_type(16))) __bf16 v16bf;
typedef __attribute__((ext_vector_type(8)))  __bf16 v8bf;
typedef __attribute__((ext_vector_type(4)))  __bf16 v4bf;
typedef __attribute__((ext_vector_type(8)))  float  v8f;

#define DIM    128
#define NHEADS 4
#define HD     32
#define NTOK   49
#define NPAD   64
#define QKSCALE 0.17677669529663687f   // 32^-0.5

// Padded LDS strides (bank-conflict avoidance; 64 banks x 4B)
#define SSTR   53                      // f32 score row stride (odd -> conflict-free)
#define SHEAD  (NPAD * SSTR)           // 3392 floats per head
#define PSTR   72                      // bf16 prob row stride (mult of 8 -> 16B-aligned frags)
#define PHEAD  (NPAD * PSTR)           // 4608 bf16 per head

// workspace layout (bytes)
#define WS_WQKV_OFF  0        // 384*128 bf16 = 98304 B
#define WS_WPROJ_OFF 98304    // 128*128 bf16 = 32768 B
#define WS_BIAS_OFF  131072   // 4*64*64 f32  = 65536 B

// ---------------------------------------------------------------------------
// Prep kernel: bf16-convert weights, gather+pad relative-position bias.
// ---------------------------------------------------------------------------
__global__ void lwa_prep_kernel(const float* __restrict__ wqkv,
                                const float* __restrict__ wproj,
                                const float* __restrict__ btab,
                                const int*   __restrict__ ridx,
                                __bf16* __restrict__ wqkv_bf,
                                __bf16* __restrict__ wproj_bf,
                                float*  __restrict__ bias_pad) {
  int i = blockIdx.x * 256 + threadIdx.x;
  if (i < 384 * 128) wqkv_bf[i]  = (__bf16)wqkv[i];
  if (i < 128 * 128) wproj_bf[i] = (__bf16)wproj[i];
  if (i < NHEADS * NPAD * NPAD) {
    int h = i >> 12, r = (i >> 6) & 63, c = i & 63;
    float v = 0.0f;
    if (r < NTOK && c < NTOK) v = btab[ridx[r * NTOK + c] * NHEADS + h];
    bias_pad[i] = v;
  }
}

// ---------------------------------------------------------------------------
// Main kernel: one window per 256-thread block (8 waves).
// ---------------------------------------------------------------------------
__global__ __launch_bounds__(256)
void lwa_attn_kernel(const float* __restrict__ x,
                     const float* __restrict__ bqkv,
                     const float* __restrict__ bproj,
                     const __bf16* __restrict__ wqkv_bf,
                     const __bf16* __restrict__ wproj_bf,
                     const float* __restrict__ bias_pad,
                     float* __restrict__ out) {
  __shared__ __align__(32) unsigned char lds[156672];
  __bf16* lds_x  = (__bf16*)(lds);            // 64x128 bf16 (x, later O)  16KB
  __bf16* lds_q  = (__bf16*)(lds + 16384);    // [4][64x32] bf16 (scaled)  16KB
  __bf16* lds_k  = (__bf16*)(lds + 32768);    // [4][64x32] bf16           16KB
  __bf16* lds_vT = (__bf16*)(lds + 49152);    // [4][32x64] bf16 (V^T)     16KB
  float*  lds_s  = (float*) (lds + 65536);    // [4][64 x SSTR] f32      53KB
  __bf16* lds_p  = (__bf16*)(lds + 119808);   // [4][64 x PSTR] bf16     36KB

  const int tid  = threadIdx.x;
  // Wave id is uniform: force into SGPR so tile indices/branches are scalar.
  const int wave = __builtin_amdgcn_readfirstlane(tid >> 5);
  const int lane = tid & 31;
  const int ln   = lane & 15;   // tile row (A) / tile col (B,C)
  const int lh   = lane >> 4;   // half select
  const long b   = blockIdx.x;
  const float* xb = x + b * (NTOK * DIM);

  // A-fragment: 16x32 bf16 tile, row-major source, contiguous k order (K = lh*16 + e), same as loadB.
  auto loadA = [&](const __bf16* base, int row, int stride, int k0) -> v16bf {
    union { v16bf v; v8bf h[2]; } u;
    const __bf16* p = base + row * stride + k0 + lh * 16;
    u.h[0] = *(const v8bf*)(p);        // e0..7  -> K = lh*16 + e
    u.h[1] = *(const v8bf*)(p + 8);    // e8..15 -> K = lh*16 + e
    return u.v;
  };
  // B-fragment: 32x16 bf16 tile; source is B^T row-major (K contiguous).
  auto loadB = [&](const __bf16* base, int col, int stride, int k0) -> v16bf {
    return *(const v16bf*)(base + col * stride + k0 + lh * 16); // K = lh*16+e
  };

  // -------- Phase 0: stage x -> LDS bf16 (float4 loads), pad rows ---------
  for (int i = tid; i < (NPAD * DIM) / 4; i += 256) {
    int r = i >> 5;                        // row = (i*4)/128
    float4 v = make_float4(0.f, 0.f, 0.f, 0.f);
    if (r < NTOK) v = ((const float4*)xb)[i];
    v4bf o; o[0] = (__bf16)v.x; o[1] = (__bf16)v.y; o[2] = (__bf16)v.z; o[3] = (__bf16)v.w;
    *(v4bf*)(lds_x + i * 4) = o;
  }
  __syncthreads();

  // -------- Phase 1: QKV = x @ Wqkv^T + b ---------------------------------
  {
    const int mt  = wave >> 1;
    const int jt0 = (wave & 1) * 12;
    v16bf a0 = loadA(lds_x, mt * 16 + ln, DIM, 0);
    v16bf a1 = loadA(lds_x, mt * 16 + ln, DIM, 32);
    v16bf a2 = loadA(lds_x, mt * 16 + ln, DIM, 64);
    v16bf a3 = loadA(lds_x, mt * 16 + ln, DIM, 96);
    for (int jj = 0; jj < 12; ++jj) {
      const int jt = jt0 + jj;             // scalar
      v8f acc = {};
      acc = __builtin_amdgcn_wmma_f32_16x16x32_bf16(false, a0, false, loadB(wqkv_bf, jt*16+ln, DIM, 0),  (short)0, acc, false, false);
      acc = __builtin_amdgcn_wmma_f32_16x16x32_bf16(false, a1, false, loadB(wqkv_bf, jt*16+ln, DIM, 32), (short)0, acc, false, false);
      acc = __builtin_amdgcn_wmma_f32_16x16x32_bf16(false, a2, false, loadB(wqkv_bf, jt*16+ln, DIM, 64), (short)0, acc, false, false);
      acc = __builtin_amdgcn_wmma_f32_16x16x32_bf16(false, a3, false, loadB(wqkv_bf, jt*16+ln, DIM, 96), (short)0, acc, false, false);
      // scalar decode: tiles never straddle part/head boundaries
      const int part = jt >> 3;            // 0=q, 1=k, 2=v
      const int head = (jt >> 1) & 3;
      const int d    = (jt & 1) * 16 + ln;
      const float bj = bqkv[jt * 16 + ln];
      if (part == 0) {
        for (int r = 0; r < 8; ++r) {
          int row = mt * 16 + lh * 8 + r;
          lds_q[head * 2048 + row * 32 + d] = (__bf16)((acc[r] + bj) * QKSCALE);
        }
      } else if (part == 1) {
        for (int r = 0; r < 8; ++r) {
          int row = mt * 16 + lh * 8 + r;
          lds_k[head * 2048 + row * 32 + d] = (__bf16)(acc[r] + bj);
        }
      } else {
        for (int r = 0; r < 8; ++r) {
          int row = mt * 16 + lh * 8 + r;
          lds_vT[head * 2048 + d * 64 + row] = (__bf16)(acc[r] + bj);
        }
      }
    }
  }
  __syncthreads();

  // -------- Phase 2: S = Qs @ K^T + bias ; K=32 ---------------------------
  {
    const int head = wave >> 1;
    const int mt0  = (wave & 1) * 2;
    const __bf16* qh = lds_q + head * 2048;
    const __bf16* kh = lds_k + head * 2048;
    v16bf a0 = loadA(qh, (mt0 + 0) * 16 + ln, HD, 0);
    v16bf a1 = loadA(qh, (mt0 + 1) * 16 + ln, HD, 0);
    const float* bh = bias_pad + head * 4096;
    for (int nt = 0; nt < 4; ++nt) {
      v16bf bm = loadB(kh, nt * 16 + ln, HD, 0);
      v8f z = {};
      v8f s0 = __builtin_amdgcn_wmma_f32_16x16x32_bf16(false, a0, false, bm, (short)0, z, false, false);
      v8f s1 = __builtin_amdgcn_wmma_f32_16x16x32_bf16(false, a1, false, bm, (short)0, z, false, false);
      int col = nt * 16 + ln;
      for (int r = 0; r < 8; ++r) {
        int row0 = (mt0 + 0) * 16 + lh * 8 + r;
        int row1 = (mt0 + 1) * 16 + lh * 8 + r;
        lds_s[head * SHEAD + row0 * SSTR + col] = s0[r] + bh[row0 * 64 + col];
        lds_s[head * SHEAD + row1 * SSTR + col] = s1[r] + bh[row1 * 64 + col];
      }
    }
  }
  __syncthreads();

  // -------- Phase 3: row softmax (f32), emit P as bf16, zero K-pad --------
  {
    int head = tid >> 6, i = tid & 63;
    float*  srow = lds_s + head * SHEAD + i * SSTR;
    __bf16* prow = lds_p + head * PHEAD + i * PSTR;
    if (i < NTOK) {
      float m = -1e30f;
      for (int j = 0; j < NTOK; ++j) m = fmaxf(m, srow[j]);
      float sum = 0.0f;
      for (int j = 0; j < NTOK; ++j) { float e = __expf(srow[j] - m); srow[j] = e; sum += e; }
      float inv = 1.0f / sum;
      for (int j = 0; j < NTOK; ++j) prow[j] = (__bf16)(srow[j] * inv);
      for (int j = NTOK; j < 64; ++j) prow[j] = (__bf16)0.0f;
    } else {
      for (int j = 0; j < 64; ++j) prow[j] = (__bf16)0.0f;
    }
  }
  __syncthreads();

  // -------- Phase 4: O = P @ V ; K=64 -------------------------------------
  {
    const int head = wave >> 1;
    const int mt0  = (wave & 1) * 2;
    const __bf16* ph = lds_p + head * PHEAD;
    const __bf16* vh = lds_vT + head * 2048;
    v16bf a00 = loadA(ph, (mt0 + 0) * 16 + ln, PSTR, 0);
    v16bf a01 = loadA(ph, (mt0 + 0) * 16 + ln, PSTR, 32);
    v16bf a10 = loadA(ph, (mt0 + 1) * 16 + ln, PSTR, 0);
    v16bf a11 = loadA(ph, (mt0 + 1) * 16 + ln, PSTR, 32);
    for (int nt = 0; nt < 2; ++nt) {
      v16bf b0 = loadB(vh, nt * 16 + ln, 64, 0);
      v16bf b1 = loadB(vh, nt * 16 + ln, 64, 32);
      v8f o0 = {}, o1 = {};
      o0 = __builtin_amdgcn_wmma_f32_16x16x32_bf16(false, a00, false, b0, (short)0, o0, false, false);
      o0 = __builtin_amdgcn_wmma_f32_16x16x32_bf16(false, a01, false, b1, (short)0, o0, false, false);
      o1 = __builtin_amdgcn_wmma_f32_16x16x32_bf16(false, a10, false, b0, (short)0, o1, false, false);
      o1 = __builtin_amdgcn_wmma_f32_16x16x32_bf16(false, a11, false, b1, (short)0, o1, false, false);
      int col = head * 32 + nt * 16 + ln;
      for (int r = 0; r < 8; ++r) {
        int row0 = (mt0 + 0) * 16 + lh * 8 + r;
        int row1 = (mt0 + 1) * 16 + lh * 8 + r;
        lds_x[row0 * 128 + col] = (__bf16)o0[r];   // reuse x buffer for O
        lds_x[row1 * 128 + col] = (__bf16)o1[r];
      }
    }
  }
  __syncthreads();

  // -------- Phase 5: Y = O @ Wproj^T + b ; K=128 --------------------------
  {
    const int mt  = wave >> 1;
    const int jt0 = (wave & 1) * 4;
    float* ob = out + b * (NTOK * DIM);
    v16bf a0 = loadA(lds_x, mt * 16 + ln, DIM, 0);
    v16bf a1 = loadA(lds_x, mt * 16 + ln, DIM, 32);
    v16bf a2 = loadA(lds_x, mt * 16 + ln, DIM, 64);
    v16bf a3 = loadA(lds_x, mt * 16 + ln, DIM, 96);
    for (int jj = 0; jj < 4; ++jj) {
      const int jt = jt0 + jj;             // scalar
      v8f acc = {};
      acc = __builtin_amdgcn_wmma_f32_16x16x32_bf16(false, a0, false, loadB(wproj_bf, jt*16+ln, DIM, 0),  (short)0, acc, false, false);
      acc = __builtin_amdgcn_wmma_f32_16x16x32_bf16(false, a1, false, loadB(wproj_bf, jt*16+ln, DIM, 32), (short)0, acc, false, false);
      acc = __builtin_amdgcn_wmma_f32_16x16x32_bf16(false, a2, false, loadB(wproj_bf, jt*16+ln, DIM, 64), (short)0, acc, false, false);
      acc = __builtin_amdgcn_wmma_f32_16x16x32_bf16(false, a3, false, loadB(wproj_bf, jt*16+ln, DIM, 96), (short)0, acc, false, false);
      int j = jt * 16 + ln;
      float bj = bproj[j];
      for (int r = 0; r < 8; ++r) {
        int row = mt * 16 + lh * 8 + r;
        if (row < NTOK) ob[row * 128 + j] = acc[r] + bj;
      }
    }
  }
}

// ---------------------------------------------------------------------------
extern "C" void kernel_launch(void* const* d_in, const int* in_sizes, int n_in,
                              void* d_out, int out_size, void* d_ws, size_t ws_size,
                              hipStream_t stream) {
  const float* x     = (const float*)d_in[0];
  // d_in[1] = q_global (unused by the reference)
  const float* wqkv  = (const float*)d_in[2];
  const float* bqkv  = (const float*)d_in[3];
  const float* wproj = (const float*)d_in[4];
  const float* bproj = (const float*)d_in[5];
  const float* btab  = (const float*)d_in[6];
  const int*   ridx  = (const int*)d_in[7];

  __bf16* wqkv_bf  = (__bf16*)((char*)d_ws + WS_WQKV_OFF);
  __bf16* wproj_bf = (__bf16*)((char*)d_ws + WS_WPROJ_OFF);
  float*  bias_pad = (float*)((char*)d_ws + WS_BIAS_OFF);

  lwa_prep_kernel<<<192, 256, 0, stream>>>(wqkv, wproj, btab, ridx,
                                           wqkv_bf, wproj_bf, bias_pad);
  lwa_attn_kernel<<<16384, 256, 0, stream>>>(x, bqkv, bproj,
                                             wqkv_bf, wproj_bf, bias_pad,
                                             (float*)d_out);
}
